// WordAttention_45956150068002
// MI455X (gfx1250) — matrix operand order repaired and checked by reading, not compiled, not run on hardware
//
#include <hip/hip_runtime.h>
#include <hip/hip_bf16.h>

// WordAttention on MI455X (gfx1250): fused GEMM+tanh+proj via bf16 WMMA
// with async LDS staging, softmax, bandwidth-bound weighted sum.

typedef __attribute__((ext_vector_type(16))) __bf16 v16bf;
typedef __attribute__((ext_vector_type(8)))  float  v8f;
typedef int v4i __attribute__((vector_size(16)));   // 128-bit payload type

#define SEQ   4096
#define BATCH 64
#define HH    256   // 2*hidden

#define AS1 __attribute__((address_space(1)))
#define AS3 __attribute__((address_space(3)))

#if __has_builtin(__builtin_amdgcn_global_load_async_to_lds_b128) && \
    __has_builtin(__builtin_amdgcn_s_wait_asynccnt)
#define USE_ASYNC 1
#else
#define USE_ASYNC 0
#endif

#if USE_ASYNC
__device__ __forceinline__ void async_b128(const unsigned int* gsrc,
                                           unsigned int* ldst) {
    __builtin_amdgcn_global_load_async_to_lds_b128(
        (AS1 v4i*)gsrc, (AS3 v4i*)ldst, 0, 0);
}
#endif

union Frag {
    unsigned int u[8];
    uint4        q[2];
    v16bf        v;
};

__device__ __forceinline__ unsigned int pack_bf2(float a, float b) {
    // round-to-nearest-even fp32 -> bf16, pack two into one dword (lo = a)
    unsigned int ua = __float_as_uint(a);
    ua = ua + 0x7FFFu + ((ua >> 16) & 1u);
    unsigned int ub = __float_as_uint(b);
    ub = ub + 0x7FFFu + ((ub >> 16) & 1u);
    return (ua >> 16) | (ub & 0xFFFF0000u);
}

__device__ __forceinline__ float fast_tanh(float x) {
#if __has_builtin(__builtin_amdgcn_tanhf)
    return __builtin_amdgcn_tanhf(x);            // v_tanh_f32
#else
    // branch-free: tanh(x) = 1 - 2/(exp(2x)+1); saturates correctly at +-inf
    float e = __expf(2.0f * x);                  // v_exp_f32
    return 1.0f - 2.0f * __builtin_amdgcn_rcpf(e + 1.0f);
#endif
}

// ---------------------------------------------------------------------------
// K0: pack W [256,256] fp32 (rows = d = K, cols = e = N) into WMMA B-fragment
// order: dword index g = ((nt*8 + kk)*32 + lane)*8 + dw
//   lane: N = nt*16 + (lane&15);  K-half = (lane>>4)*16
//   dw:   K pair = kk*32 + Khalf + 2*dw, 2*dw+1
// ---------------------------------------------------------------------------
__global__ void __launch_bounds__(256) pack_w_kernel(
    const float* __restrict__ W, unsigned int* __restrict__ wsb)
{
    const int g    = blockIdx.x * 256 + threadIdx.x;   // 0..32767
    const int dw   = g & 7;
    const int lane = (g >> 3) & 31;
    const int kk   = (g >> 8) & 7;
    const int nt   = g >> 11;
    const int n    = nt * 16 + (lane & 15);
    const int k    = kk * 32 + (lane >> 4) * 16 + dw * 2;
    wsb[g] = pack_bf2(W[(size_t)k * HH + n], W[(size_t)(k + 1) * HH + n]);
}

// ---------------------------------------------------------------------------
// K1: fused GEMM + bias + tanh + proj-dot. Grid: 2048 blocks x 256 thr.
// Each wave owns a 16-row M-tile (row = s*64+b). Raw scores written
// transposed into the attn region of d_out as [b][s].
// ---------------------------------------------------------------------------
__global__ void __launch_bounds__(256) gemm_score_kernel(
    const float* __restrict__ enc, const unsigned int* __restrict__ wsb,
    const float* __restrict__ bias, const float* __restrict__ proj,
    float* __restrict__ attn)
{
    __shared__ unsigned int Bs[2][2048];       // 2 x 8 KB B N-tile buffers

    const int t    = threadIdx.x;
    const int lane = t & 31;
    const int wave = t >> 5;
    const int col  = lane & 15;                // N%16 for B/C; M for A
    const int half = lane >> 4;
    const int tileBase = blockIdx.x * 128 + wave * 16;
    const int row  = tileBase + col;
    const float* rp = enc + (size_t)row * HH;

    // Preload + convert A fragments (16 rows x 256 K) to bf16 register layout.
    // This code's A order (B packed to match): lanes 0-15 hold K {0..15}; lanes 16-31 {16..31}.
    Frag a[8];
#pragma unroll
    for (int kk = 0; kk < 8; ++kk) {
        const int k0 = kk * 32 + half * 16;
        float4 f0 = *(const float4*)(rp + k0);
        float4 f1 = *(const float4*)(rp + k0 + 4);
        float4 f2 = *(const float4*)(rp + k0 + 8);
        float4 f3 = *(const float4*)(rp + k0 + 12);
        a[kk].u[0] = pack_bf2(f0.x, f0.y);
        a[kk].u[1] = pack_bf2(f0.z, f0.w);
        a[kk].u[2] = pack_bf2(f1.x, f1.y);
        a[kk].u[3] = pack_bf2(f1.z, f1.w);
        a[kk].u[4] = pack_bf2(f2.x, f2.y);
        a[kk].u[5] = pack_bf2(f2.z, f2.w);
        a[kk].u[6] = pack_bf2(f3.x, f3.y);
        a[kk].u[7] = pack_bf2(f3.z, f3.w);
    }

    float p[8];
#pragma unroll
    for (int v = 0; v < 8; ++v) p[v] = 0.f;

#if USE_ASYNC
    // Prologue: async-stage N-tile 0 into buffer 0 (2 x b128 per thread).
    async_b128(wsb + t * 8,     &Bs[0][t * 8]);
    async_b128(wsb + t * 8 + 4, &Bs[0][t * 8 + 4]);
    __builtin_amdgcn_s_wait_asynccnt(0);
    __syncthreads();
#endif

    for (int nt = 0; nt < 16; ++nt) {
#if USE_ASYNC
        const unsigned int* bs = Bs[nt & 1];
        if (nt + 1 < 16) {                     // overlap next tile's DMA
            const unsigned int* src = wsb + (nt + 1) * 2048 + t * 8;
            unsigned int*       dst = &Bs[(nt + 1) & 1][t * 8];
            async_b128(src,     dst);
            async_b128(src + 4, dst + 4);
        }
#else
        const unsigned int* bs = Bs[0];
        __syncthreads();
        {   // cooperative synchronous stage: 2048 dwords, 2 uint4 per thread
            const uint4* src = (const uint4*)wsb + nt * 512 + t * 2;
            uint4* dst = (uint4*)Bs[0] + t * 2;
            dst[0] = src[0];
            dst[1] = src[1];
        }
        if (nt + 1 < 16) __builtin_prefetch(wsb + (nt + 1) * 2048, 0, 1);
        __syncthreads();
#endif

        const float bv = bias[nt * 16 + col];
        const float pv = proj[nt * 16 + col];

        v8f acc = {0.f, 0.f, 0.f, 0.f, 0.f, 0.f, 0.f, 0.f};
#pragma unroll
        for (int kk = 0; kk < 8; ++kk) {
            Frag bfr;
            bfr.q[0] = *(const uint4*)&bs[(kk * 32 + lane) * 8];
            bfr.q[1] = *(const uint4*)&bs[(kk * 32 + lane) * 8 + 4];
            acc = __builtin_amdgcn_wmma_f32_16x16x32_bf16(
                false, a[kk].v, false, bfr.v, (short)0, acc, false, false);
        }
        // epilogue: C layout -> lane holds (m = v + half*8, n = nt*16 + col)
#pragma unroll
        for (int v = 0; v < 8; ++v)
            p[v] += fast_tanh(acc[v] + bv) * pv;

#if USE_ASYNC
        __builtin_amdgcn_s_wait_asynccnt(0);
        __syncthreads();
#endif
    }

    // reduce over the 16 lanes of each half (sum over n for fixed row m)
#pragma unroll
    for (int v = 0; v < 8; ++v) {
        p[v] += __shfl_xor(p[v], 1, 32);
        p[v] += __shfl_xor(p[v], 2, 32);
        p[v] += __shfl_xor(p[v], 4, 32);
        p[v] += __shfl_xor(p[v], 8, 32);
    }
    if (col == 0) {                            // lanes 0 and 16
        const int rbase = tileBase + half * 8;
#pragma unroll
        for (int v = 0; v < 8; ++v) {
            const int r = rbase + v;           // r = s*64 + b
            attn[(size_t)(r & 63) * SEQ + (r >> 6)] = p[v];
        }
    }
}

// ---------------------------------------------------------------------------
// K2: per-batch softmax over seq (in place on [b][4096]); also zeroes vectors.
// ---------------------------------------------------------------------------
__global__ void __launch_bounds__(256) softmax_kernel(float* __restrict__ out)
{
    const int b = blockIdx.x, t = threadIdx.x;
    out[b * HH + t] = 0.f;                     // zero vectors region [64][256]
    float* attn = out + BATCH * HH + (size_t)b * SEQ;

    __shared__ float red[8];
    const int lane = t & 31, wv = t >> 5;

    float mx = -3.402823466e38f;
    for (int s = t; s < SEQ; s += 256) mx = fmaxf(mx, attn[s]);
#pragma unroll
    for (int m = 16; m >= 1; m >>= 1) mx = fmaxf(mx, __shfl_xor(mx, m, 32));
    if (lane == 0) red[wv] = mx;
    __syncthreads();
    mx = red[0];
#pragma unroll
    for (int i = 1; i < 8; ++i) mx = fmaxf(mx, red[i]);
    __syncthreads();

    float sum = 0.f;
    for (int s = t; s < SEQ; s += 256) sum += __expf(attn[s] - mx);
#pragma unroll
    for (int m = 16; m >= 1; m >>= 1) sum += __shfl_xor(sum, m, 32);
    if (lane == 0) red[wv] = sum;
    __syncthreads();
    sum = 0.f;
#pragma unroll
    for (int i = 0; i < 8; ++i) sum += red[i];
    const float inv = 1.f / sum;

    for (int s = t; s < SEQ; s += 256)
        attn[s] = __expf(attn[s] - mx) * inv;
}

// ---------------------------------------------------------------------------
// K3: vectors[b][d] = sum_s attn[b][s] * enc[s][b][d]. Grid (64 b, 16 chunks).
// ---------------------------------------------------------------------------
__global__ void __launch_bounds__(256) weighted_sum_kernel(
    const float* __restrict__ enc, float* __restrict__ out)
{
    const int b = blockIdx.x, ck = blockIdx.y, t = threadIdx.x;
    const float* attn = out + BATCH * HH + (size_t)b * SEQ + ck * 256;

    __shared__ float wsh[256];
    wsh[t] = attn[t];
    __syncthreads();

    const int s0 = ck * 256;
    float acc = 0.f;
#pragma unroll 4
    for (int i = 0; i < 256; ++i)
        acc += wsh[i] * enc[((size_t)(s0 + i) * BATCH + b) * HH + t];

    unsafeAtomicAdd(&out[b * HH + t], acc);
}

// ---------------------------------------------------------------------------
extern "C" void kernel_launch(void* const* d_in, const int* in_sizes, int n_in,
                              void* d_out, int out_size, void* d_ws, size_t ws_size,
                              hipStream_t stream) {
    const float* enc  = (const float*)d_in[0];  // [4096,64,256]
    const float* W    = (const float*)d_in[1];  // [256,256]
    const float* bias = (const float*)d_in[2];  // [256,1]
    const float* proj = (const float*)d_in[3];  // [256,1]
    float* out = (float*)d_out;                 // [64*256 vectors | 64*4096 attn]
    unsigned int* wsb = (unsigned int*)d_ws;    // 128 KB packed bf16 W

    pack_w_kernel<<<128, 256, 0, stream>>>(W, wsb);
    gemm_score_kernel<<<2048, 256, 0, stream>>>(enc, wsb, bias, proj,
                                                out + BATCH * HH);
    softmax_kernel<<<BATCH, 256, 0, stream>>>(out);
    weighted_sum_kernel<<<dim3(BATCH, 16), 256, 0, stream>>>(enc, out);
}
